// DecayBlock_6786048327776
// MI455X (gfx1250) — matrix operand order repaired and checked by reading, not compiled, not run on hardware
//
#include <hip/hip_runtime.h>
#include <stdint.h>

// ---- problem constants (from reference) ----
#define BB 8
#define SS 4096
#define DM 1024
#define HH 16
#define DCONV 4
#define CHK 64
#define NCH (SS / CHK)     // 64 chunks
#define DH (DM / HH)       // 64 head dim
#define NP (DM + HH)       // 1040 = in_proj output width

typedef __attribute__((ext_vector_type(16))) __bf16 v16bf;
typedef __attribute__((ext_vector_type(8)))  float  v8f;

// ---- bf16 helpers (round-to-nearest-even) ----
__device__ __forceinline__ __bf16 f2bf(float f) {
  unsigned u = __builtin_bit_cast(unsigned, f);
  unsigned r = u + 0x7FFFu + ((u >> 16) & 1u);
  unsigned short h = (unsigned short)(r >> 16);
  return __builtin_bit_cast(__bf16, h);
}
__device__ __forceinline__ float bf2f(__bf16 b) {
  unsigned short h = __builtin_bit_cast(unsigned short, b);
  unsigned u = ((unsigned)h) << 16;
  return __builtin_bit_cast(float, u);
}

// low 32 bits of a generic pointer to LDS == LDS byte address (aperture in high word)
__device__ __forceinline__ unsigned lds_addr32(const void* p) {
  return (unsigned)(uintptr_t)p;
}

// async Global->LDS 16-byte copy, GVS mode: mem = saddr + voffset; lds = vdst
__device__ __forceinline__ void async_copy_b128(unsigned lds, unsigned voff,
                                                const void* sbase) {
  asm volatile("global_load_async_to_lds_b128 %0, %1, %2"
               :
               : "v"(lds), "v"(voff), "s"(sbase)
               : "memory");
}
__device__ __forceinline__ void wait_async0() {
  asm volatile("s_wait_asynccnt 0x0" ::: "memory");
}

// ---- WMMA fragment loads (16-bit, wave32; contiguous k order, same for A and B) ----
// A 16x32: lanes 0-15 = rows, K 0..15; lanes 16-31 same rows, K 16..31
__device__ __forceinline__ v16bf ld_a_frag(const __bf16* p0, const __bf16* p1) {
  v16bf r;
#pragma unroll
  for (int i = 0; i < 8; ++i) { r[i] = p0[i]; r[i + 8] = p1[i]; }
  return r;
}
// B 32x16: lanes 0-15 = cols with K 0..15; lanes 16-31 = cols with K 16..31
__device__ __forceinline__ v16bf ld_b_frag(const __bf16* p) {
  v16bf r;
#pragma unroll
  for (int i = 0; i < 16; ++i) r[i] = p[i];
  return r;
}
__device__ __forceinline__ v8f wmma_bf16(v16bf a, v16bf b, v8f c) {
  return __builtin_amdgcn_wmma_f32_16x16x32_bf16(false, a, false, b, (short)0, c,
                                                 false, false);
}

// ---------------------------------------------------------------------------
// Kernel 1: fp32 -> bf16 conversion (weights)
// ---------------------------------------------------------------------------
__global__ void f2bf_kernel(const float* __restrict__ in, __bf16* __restrict__ out,
                            int n) {
  int i = blockIdx.x * 256 + threadIdx.x;
  if (i < n) out[i] = f2bf(in[i]);
}

// ---------------------------------------------------------------------------
// Kernel 2/7: C[m,n] = sum_k A[m,k] * Bw[n,k]   (A fp32 or bf16; Bw bf16; C fp32)
// Block tile 128x64, 8 wave32 waves in 4(M)x2(N) grid, each wave 32x32.
// B panel staged with async Global->LDS; A panel async too when already bf16.
// ---------------------------------------------------------------------------
template <bool A_F32>
__global__ __launch_bounds__(256) void gemm_wmma_kernel(
    const void* __restrict__ Ap, const __bf16* __restrict__ Bw,
    float* __restrict__ Cp, int M, int N, int K) {
  constexpr int BM = 128, BN = 64, BK = 32;
  constexpr int LDA = BK + 4;          // halves; pad against bank conflicts
  constexpr int LDAB = LDA * 2;        // 72 bytes per LDS row
  __shared__ __bf16 As[BM * LDA];
  __shared__ __bf16 Bs[BN * LDA];

  const int tid = threadIdx.x;
  const int lane = tid & 31;
  const int wave = tid >> 5;
  const int wm = wave >> 1;  // 0..3
  const int wn = wave & 1;   // 0..1
  const int hl = lane >> 4;  // half-wave
  const int l16 = lane & 15;
  const int bm = blockIdx.x * BM;
  const int bn = blockIdx.y * BN;

  const unsigned as_base = lds_addr32(&As[0]);
  const unsigned bs_base = lds_addr32(&Bs[0]);

  v8f acc[2][2] = {};

  for (int k0 = 0; k0 < K; k0 += BK) {
    __syncthreads();
    // ---- stage A ----
    if constexpr (A_F32) {
      // fp32 -> bf16 convert through VGPRs
      for (int i = tid; i < BM * BK; i += 256) {
        int r = i >> 5, cc = i & 31;
        int gm = bm + r; if (gm >= M) gm = M - 1;
        float v = ((const float*)Ap)[(size_t)gm * K + k0 + cc];
        As[r * LDA + cc] = f2bf(v);
      }
      if (k0 + BK < K) {
        int gm = bm + (tid & 127); if (gm >= M) gm = M - 1;
        __builtin_prefetch((const char*)Ap + ((size_t)gm * K + k0 + BK) * 4, 0, 1);
      }
    } else {
      // already bf16: async copy, 16B chunks (4 per 64B row of the K panel)
      const void* abase = (const char*)Ap + (size_t)k0 * 2;
#pragma unroll
      for (int it = 0; it < 2; ++it) {
        int i = tid + it * 256;               // 512 chunks total
        int r = i >> 2, ch = i & 3;
        int gm = bm + r; if (gm >= M) gm = M - 1;
        unsigned voff = (unsigned)((size_t)gm * K * 2 + ch * 16);
        async_copy_b128(as_base + (unsigned)(r * LDAB + ch * 16), voff, abase);
      }
    }
    // ---- stage B (always bf16): async copy, 256 chunks / 256 threads ----
    {
      const void* bbase = (const char*)Bw + (size_t)k0 * 2;
      int r = tid >> 2, ch = tid & 3;
      int gn = bn + r; if (gn >= N) gn = N - 1;
      unsigned voff = (unsigned)((size_t)gn * K * 2 + ch * 16);
      async_copy_b128(bs_base + (unsigned)(r * LDAB + ch * 16), voff, bbase);
    }
    wait_async0();
    __syncthreads();

    v16bf af[2], bfr[2];
#pragma unroll
    for (int tm = 0; tm < 2; ++tm) {
      const __bf16* ap = &As[(wm * 32 + tm * 16 + l16) * LDA + hl * 16];
      af[tm] = ld_a_frag(ap, ap + 8);
    }
#pragma unroll
    for (int tn = 0; tn < 2; ++tn)
      bfr[tn] = ld_b_frag(&Bs[(wn * 32 + tn * 16 + l16) * LDA + hl * 16]);
#pragma unroll
    for (int tm = 0; tm < 2; ++tm)
#pragma unroll
      for (int tn = 0; tn < 2; ++tn)
        acc[tm][tn] = wmma_bf16(af[tm], bfr[tn], acc[tm][tn]);
  }

  // epilogue: C/D layout: VGPR r -> M = r (lanes 0-15) / 8+r (lanes 16-31), N = lane&15
#pragma unroll
  for (int tm = 0; tm < 2; ++tm) {
#pragma unroll
    for (int tn = 0; tn < 2; ++tn) {
      int col = bn + wn * 32 + tn * 16 + l16;
      if (col < N) {
#pragma unroll
        for (int r = 0; r < 8; ++r) {
          int row = bm + wm * 32 + tm * 16 + (hl ? 8 + r : r);
          if (row < M) Cp[(size_t)row * N + col] = acc[tm][tn][r];
        }
      }
    }
  }
}

// ---------------------------------------------------------------------------
// Kernel 3: depthwise causal conv (width 4) + A = w * w_base
// ---------------------------------------------------------------------------
__global__ __launch_bounds__(256) void conv_a_kernel(
    const float* __restrict__ xw, const float* __restrict__ conv_w,
    const float* __restrict__ conv_b, const float* __restrict__ w_base,
    __bf16* __restrict__ xh, float* __restrict__ Aarr) {
  int bs = blockIdx.x;            // b*S + s
  int s = bs & (SS - 1);
  int d = blockIdx.y * 256 + threadIdx.x;
  float acc = conv_b[d];
#pragma unroll
  for (int j = 0; j < DCONV; ++j) {
    int sj = s - (DCONV - 1) + j;
    if (sj >= 0) acc += conv_w[d * DCONV + j] * xw[((size_t)bs - (s - sj)) * NP + d];
  }
  xh[(size_t)bs * DM + d] = f2bf(acc);
  if (blockIdx.y == 0 && threadIdx.x < HH) {
    int h = threadIdx.x;
    Aarr[(size_t)bs * HH + h] = xw[(size_t)bs * NP + DM + h] * w_base[h];
  }
}

// ---------------------------------------------------------------------------
// Kernel 4: per-chunk cumsum(A), decay-state weights, chunk states
// ---------------------------------------------------------------------------
__global__ __launch_bounds__(64) void chunk_states_kernel(
    const float* __restrict__ Aarr, const __bf16* __restrict__ xh,
    float* __restrict__ cumA, float* __restrict__ states,
    float* __restrict__ normdecay) {
  int blk = blockIdx.x;
  int c = blk % NCH;
  int bh = blk / NCH;
  int h = bh % HH, b = bh / HH;
  __shared__ float ca[CHK];
  __shared__ float wts[CHK];
  __shared__ float s_mn;
  __shared__ __bf16 xs[CHK * DH];
  int t = threadIdx.x;
  int s0 = c * CHK;

  ca[t] = Aarr[((size_t)(b * SS + s0 + t)) * HH + h];
  const __bf16* xrow = &xh[((size_t)(b * SS + s0 + t)) * DM + h * DH];
#pragma unroll 8
  for (int n = 0; n < DH; ++n) xs[t * DH + n] = xrow[n];
  __syncthreads();
  if (t == 0) {
    float run = 0.f, mn = 1e30f;
    for (int i = 0; i < CHK; ++i) {
      run += ca[i];
      ca[i] = run;
      mn = fminf(mn, run);
    }
    s_mn = mn;
  }
  __syncthreads();
  float w = __expf(s_mn - ca[t]);  // exp((cA_last - cA[l]) - max_l(...))
  wts[t] = w;
  cumA[((size_t)bh * NCH + c) * CHK + t] = ca[t];
  __syncthreads();
  float acc = 0.f;
#pragma unroll 8
  for (int l = 0; l < CHK; ++l) acc += wts[l] * bf2f(xs[l * DH + t]);
  states[((size_t)bh * NCH + c) * DH + t] = acc;
  if (t == 0) {
    float nd = 0.f;
    for (int l = 0; l < CHK; ++l) nd += wts[l];
    normdecay[(size_t)bh * NCH + c] = nd;
  }
}

// ---------------------------------------------------------------------------
// Kernel 5: inter-chunk combine (65x65 decay matrix per (b,h))
// ---------------------------------------------------------------------------
__global__ __launch_bounds__(64) void interchunk_kernel(
    const float* __restrict__ cumA, const float* __restrict__ states,
    const float* __restrict__ normdecay, float* __restrict__ statesprev,
    float* __restrict__ normprev) {
  int bh = blockIdx.x;  // B*H
  __shared__ float cc[NCH + 1];
  __shared__ float rm[NCH + 1];
  __shared__ float nd[NCH];
  __shared__ float st[NCH * DH];        // 16 KB
  __shared__ float dc[NCH * (NCH + 1)]; // 16.6 KB
  int t = threadIdx.x;

  for (int i = t; i < NCH * DH; i += 64)
    st[i] = states[(size_t)bh * NCH * DH + i];
  if (t < NCH) nd[t] = normdecay[(size_t)bh * NCH + t];
  if (t == 0) {
    cc[0] = 0.f;
    for (int z = 1; z <= NCH; ++z)
      cc[z] = cc[z - 1] + cumA[((size_t)bh * NCH + (z - 1)) * CHK + (CHK - 1)];
    float mnp = cc[0];
    for (int z = 0; z <= NCH; ++z) {
      mnp = fminf(mnp, cc[z]);
      rm[z] = cc[z] - mnp;
    }
  }
  __syncthreads();
  for (int i = t; i < NCH * (NCH + 1); i += 64) {
    int z = i / (NCH + 1), cx = i % (NCH + 1);
    dc[i] = (cx >= 1 && cx <= z) ? __expf(cc[z] - cc[cx] - rm[z]) : 0.f;
  }
  __syncthreads();
  for (int z = 0; z < NCH; ++z) {
    float acc = 0.f;
    for (int cx = 1; cx <= z; ++cx) acc += dc[z * (NCH + 1) + cx] * st[(cx - 1) * DH + t];
    statesprev[((size_t)bh * NCH + z) * DH + t] = acc;
  }
  {
    int z = t;
    float acc = 0.f;
    for (int cx = 1; cx <= z; ++cx) acc += dc[z * (NCH + 1) + cx] * nd[cx - 1];
    normprev[(size_t)bh * NCH + z] = acc;
  }
}

// ---------------------------------------------------------------------------
// Kernel 6: per-chunk SSD output: Y = L@X (WMMA) + states_prev (x) sdo; * rcp(norm)
// ---------------------------------------------------------------------------
__global__ __launch_bounds__(128) void ssd_out_kernel(
    const __bf16* __restrict__ xh, const float* __restrict__ cumA,
    const float* __restrict__ statesprev, const float* __restrict__ normprev,
    __bf16* __restrict__ yout) {
  constexpr int LDL = CHK + 8;  // 72 halves
  int blk = blockIdx.x;
  int c = blk % NCH;
  int bh = blk / NCH;
  int h = bh % HH, b = bh / HH;
  __shared__ float ca[CHK], rmax[CHK], ndiag[CHK], sdo[CHK], sprev[DH];
  __shared__ float s_camax;
  __shared__ __bf16 Ls[CHK * LDL];
  __shared__ __bf16 Xs[DH * LDL];  // transposed: [n][s]
  int t = threadIdx.x;

  if (t < CHK) ca[t] = cumA[((size_t)bh * NCH + c) * CHK + t];
  if (t >= 64 && t < 128)
    sprev[t - 64] = statesprev[((size_t)bh * NCH + c) * DH + (t - 64)];
  __syncthreads();
  if (t == 0) {
    float mnp = ca[0], mx = ca[0];
    for (int l = 0; l < CHK; ++l) {
      mnp = fminf(mnp, ca[l]);
      rmax[l] = ca[l] - mnp;
      mx = fmaxf(mx, ca[l]);
    }
    s_camax = mx;
  }
  __syncthreads();
  if (t < CHK) {
    sdo[t] = __expf(ca[t] - s_camax);
    float sum = 0.f;
    for (int s = 0; s <= t; ++s) sum += __expf(ca[t] - ca[s] - rmax[t]);
    ndiag[t] = sum;
  }
  for (int i = t; i < CHK * CHK; i += 128) {
    int l = i >> 6, s = i & 63;
    float v = (s <= l) ? __expf(ca[l] - ca[s] - rmax[l]) : 0.f;
    Ls[l * LDL + s] = f2bf(v);
  }
  for (int i = t; i < CHK * DH; i += 128) {
    int s = i >> 6, n = i & 63;
    Xs[n * LDL + s] = xh[((size_t)(b * SS + c * CHK + s)) * DM + h * DH + n];
  }
  __syncthreads();

  int lane = t & 31, hl = lane >> 4, l16 = lane & 15;
  int w = t >> 5;
  v8f acc[4] = {};
#pragma unroll
  for (int k0 = 0; k0 < CHK; k0 += 32) {
    const __bf16* ap = &Ls[(w * 16 + l16) * LDL + k0 + hl * 16];
    v16bf af = ld_a_frag(ap, ap + 8);
#pragma unroll
    for (int tn = 0; tn < 4; ++tn) {
      v16bf bfr = ld_b_frag(&Xs[(tn * 16 + l16) * LDL + k0 + hl * 16]);
      acc[tn] = wmma_bf16(af, bfr, acc[tn]);
    }
  }

  float np = normprev[(size_t)bh * NCH + c];
#pragma unroll
  for (int tn = 0; tn < 4; ++tn) {
    int n = tn * 16 + l16;
    float sp = sprev[n];
#pragma unroll
    for (int r = 0; r < 8; ++r) {
      int l = w * 16 + (hl ? 8 + r : r);
      float dec = sdo[l];
      float Y = acc[tn][r] + sp * dec;
      float nrm = ndiag[l] + np * dec;
      float inv = __builtin_amdgcn_rcpf(nrm);  // v_rcp_f32, no div expansion
      yout[((size_t)(b * SS + c * CHK + l)) * DM + h * DH + n] = f2bf(Y * inv);
    }
  }
}

// ---------------------------------------------------------------------------
// Host launcher
// ---------------------------------------------------------------------------
extern "C" void kernel_launch(void* const* d_in, const int* in_sizes, int n_in,
                              void* d_out, int out_size, void* d_ws, size_t ws_size,
                              hipStream_t stream) {
  const float* x        = (const float*)d_in[0];  // (B,S,DM)
  const float* w1       = (const float*)d_in[1];  // (NP,DM)
  const float* conv_w   = (const float*)d_in[2];  // (DM,1,4)
  const float* conv_b   = (const float*)d_in[3];  // (DM)
  const float* w_base   = (const float*)d_in[4];  // (H)
  const float* w2       = (const float*)d_in[5];  // (DM,DM)
  float* out            = (float*)d_out;          // (B,S,DM)

  const size_t Mrows = (size_t)BB * SS;  // 32768

  char* ws = (char*)d_ws;
  size_t off = 0;
  auto carve = [&](size_t bytes) {
    size_t o = off;
    off = (off + bytes + 255) & ~(size_t)255;
    return o;
  };
  __bf16* w1b   = (__bf16*)(ws + carve((size_t)NP * DM * 2));
  __bf16* w2b   = (__bf16*)(ws + carve((size_t)DM * DM * 2));
  float*  xw    = (float*) (ws + carve(Mrows * NP * 4));
  __bf16* xhb   = (__bf16*)(ws + carve(Mrows * DM * 2));
  float*  Aarr  = (float*) (ws + carve(Mrows * HH * 4));
  float*  cumA  = (float*) (ws + carve((size_t)BB * HH * NCH * CHK * 4));
  float*  stat  = (float*) (ws + carve((size_t)BB * HH * NCH * DH * 4));
  float*  ndec  = (float*) (ws + carve((size_t)BB * HH * NCH * 4));
  float*  sprev = (float*) (ws + carve((size_t)BB * HH * NCH * DH * 4));
  float*  nprev = (float*) (ws + carve((size_t)BB * HH * NCH * 4));
  __bf16* yb    = (__bf16*)(ws + carve(Mrows * DM * 2));
  (void)ws_size; (void)in_sizes; (void)n_in; (void)out_size;

  {
    int n1 = NP * DM;
    f2bf_kernel<<<(n1 + 255) / 256, 256, 0, stream>>>(w1, w1b, n1);
    int n2 = DM * DM;
    f2bf_kernel<<<(n2 + 255) / 256, 256, 0, stream>>>(w2, w2b, n2);
  }
  {
    dim3 grid((unsigned)(Mrows / 128), (NP + 63) / 64);
    gemm_wmma_kernel<true><<<grid, 256, 0, stream>>>(x, w1b, xw,
                                                     (int)Mrows, NP, DM);
  }
  {
    dim3 grid((unsigned)Mrows, DM / 256);
    conv_a_kernel<<<grid, 256, 0, stream>>>(xw, conv_w, conv_b, w_base, xhb, Aarr);
  }
  chunk_states_kernel<<<BB * HH * NCH, 64, 0, stream>>>(Aarr, xhb, cumA, stat, ndec);
  interchunk_kernel<<<BB * HH, 64, 0, stream>>>(cumA, stat, ndec, sprev, nprev);
  ssd_out_kernel<<<BB * HH * NCH, 128, 0, stream>>>(xhb, cumA, sprev, nprev, yb);
  {
    dim3 grid((unsigned)(Mrows / 128), DM / 64);
    gemm_wmma_kernel<false><<<grid, 256, 0, stream>>>(yb, w2b, out,
                                                      (int)Mrows, DM, DM);
  }
}
